// MessagePassingModel_6493990552148
// MI455X (gfx1250) — matrix operand order repaired and checked by reading, not compiled, not run on hardware
//
#include <hip/hip_runtime.h>
#include <hip/hip_bf16.h>
#include <math.h>

#define NN 50000
#define EE 400000
#define FF 32
#define KK 16
#define BB 10
#define CUTR 10.0f
#define EPSR 1e-8f

typedef __attribute__((ext_vector_type(16))) _Float16 v16h;
typedef __attribute__((ext_vector_type(8)))  _Float16 v8h;
typedef __attribute__((ext_vector_type(2)))  _Float16 v2h;
typedef __attribute__((ext_vector_type(8)))  float    v8f;

__device__ __constant__ float c_binom[16] = {
  1.f,15.f,105.f,455.f,1365.f,3003.f,5005.f,6435.f,
  6435.f,5005.f,3003.f,1365.f,455.f,105.f,15.f,1.f };

static inline int cdiv(long a, long b) { return (int)((a + b - 1) / b); }

__device__ inline void fatomic(float* p, float v) { unsafeAtomicAdd(p, v); }

// ---------------- elementwise utils ----------------
__global__ void k_zero(float* __restrict__ p, long n) {
  long i = (long)blockIdx.x * blockDim.x + threadIdx.x;
  if (i < n) p[i] = 0.f;
}

__global__ void k_add_inplace(float* __restrict__ dst, const float* __restrict__ src, long n) {
  long i = (long)blockIdx.x * blockDim.x + threadIdx.x;
  if (i < n) dst[i] += src[i];
}

// ---------------- radial basis + harmonics ----------------
__global__ void k_basis(const float* __restrict__ pos,
                        const int* __restrict__ dst, const int* __restrict__ src,
                        float* __restrict__ b0, float* __restrict__ b1,
                        float* __restrict__ uvec, float* __restrict__ rvec) {
  int e = blockIdx.x * blockDim.x + threadIdx.x;
  if (e >= EE) return;
  int sn = src[e], dn = dst[e];
  float d0 = pos[sn*3+0] - pos[dn*3+0];
  float d1 = pos[sn*3+1] - pos[dn*3+1];
  float d2 = pos[sn*3+2] - pos[dn*3+2];
  float r  = sqrtf(d0*d0 + d1*d1 + d2*d2 + EPSR);
  float t  = r / (1.f + r);
  float omt = 1.f - t;
  float cut = 0.f;
  if (r < CUTR) {
    float denom = (CUTR - r) * (CUTR + r);
    cut = expf(-r * r / denom);
  }
  float inv_r = 1.f / r;
  float u0 = d0 * inv_r, u1 = d1 * inv_r, u2 = d2 * inv_r;
  rvec[e] = r;
  uvec[e*3+0] = u0; uvec[e*3+1] = u1; uvec[e*3+2] = u2;
  float o[16];
  o[0] = 1.f;
  #pragma unroll
  for (int j = 1; j < 16; ++j) o[j] = o[j-1] * omt;
  float tp = 1.f;
  #pragma unroll
  for (int k = 0; k < 16; ++k) {
    float rad = c_binom[k] * tp * o[15-k] * cut;
    tp *= t;
    b0[(long)e*KK + k] = rad;
    b1[((long)e*3 + 0)*KK + k] = u0 * rad;
    b1[((long)e*3 + 1)*KK + k] = u1 * rad;
    b1[((long)e*3 + 2)*KK + k] = u2 * rad;
  }
}

// ---------------- embedding ----------------
__global__ void k_embed(const int* __restrict__ Z, const float* __restrict__ embed,
                        float* __restrict__ s) {
  long i = (long)blockIdx.x * blockDim.x + threadIdx.x;
  if (i >= (long)NN * FF) return;
  int n = (int)(i >> 5), f = (int)(i & 31);
  s[i] = embed[Z[n]*FF + f];
}

// ---------------- WMMA GEMM:  OUT[M x 32] = IN[M x KD] @ W[KD x 32] (+bias)(+resid) ----
// 256 threads = 8 waves; wave w handles row-tile (blockIdx*8 + w) of 16 rows.
// A and W staged in LDS as f16 in per-lane-contiguous order so operands load as
// ds_load_b128s. All M are multiples of 16 => only wave-uniform guards (EXEC
// stays all-ones for WMMA).
template<int KD, bool HAS_BIAS, bool HAS_RESID>
__global__ void k_gemm32(const float* __restrict__ in, const float* __restrict__ W,
                         const float* __restrict__ bias, const float* __restrict__ resid,
                         float* __restrict__ out, int M) {
  __shared__ __attribute__((aligned(16))) _Float16 Wt[32][32];     // [n][k] transposed
  __shared__ __attribute__((aligned(16))) _Float16 Ah[8][16][32];  // [row][k] per wave
  int tid  = threadIdx.x;
  int wave = tid >> 5;
  int lane = tid & 31;

  // stage W transposed (zero-pad K beyond KD)
  for (int i = tid; i < 32*32; i += 256) {
    int kk = i >> 5, nn = i & 31;
    Wt[nn][kk] = (kk < KD) ? (_Float16)W[kk*32 + nn] : (_Float16)0.f;
  }

  long row0 = ((long)blockIdx.x * 8 + wave) * 16;
  bool active = row0 < (long)M;            // wave-uniform
  if (active) {
    // 256 f16-pairs: coalesced float2 global loads -> packed f16 LDS stores
    for (int p = lane; p < 256; p += 32) {
      int el  = p * 2;
      int row = el >> 5, col = el & 31;    // col is even; per-lane constant
      v2h hv;
      if (col < KD) {
        const float2* gp = (const float2*)(in + (row0 + row)*(long)KD + col);
        float2 d = *gp;
        hv[0] = (_Float16)d.x; hv[1] = (_Float16)d.y;
      } else {
        hv[0] = (_Float16)0.f; hv[1] = (_Float16)0.f;
      }
      *(v2h*)&Ah[wave][row][col] = hv;
    }
  }
  __syncthreads();
  if (!active) return;

  // A operand, contiguous K order (matches B):
  // lanes 0-15: K 0..15 ; lanes 16-31: K 16..31
  int rloc  = lane & 15;
  int khalf = lane >> 4;
  int kb1   = khalf * 16;
  v8h alo = *(const v8h*)&Ah[wave][rloc][kb1];        // ds_load_b128
  v8h ahi = *(const v8h*)&Ah[wave][rloc][kb1 + 8];    // ds_load_b128
  v16h a  = __builtin_shufflevector(alo, ahi,
              0,1,2,3,4,5,6,7,8,9,10,11,12,13,14,15);

  #pragma unroll
  for (int ctile = 0; ctile < 2; ++ctile) {
    int col = ctile*16 + (lane & 15);
    int kb  = khalf * 16;
    v16h b = *(const v16h*)&Wt[col][kb];               // 2x ds_load_b128
    v8f c;
    float binit = HAS_BIAS ? bias[col] : 0.f;
    #pragma unroll
    for (int j = 0; j < 8; ++j) c[j] = binit;
    c = __builtin_amdgcn_wmma_f32_16x16x32_f16(
        false, a, false, b, (short)0, c, false, false);
    // base pointer once; stores use immediate offsets (row stride 32 floats)
    float*       po = out + (row0 + khalf*8)*32 + col;
    const float* pr = HAS_RESID ? (resid + (row0 + khalf*8)*32 + col) : nullptr;
    #pragma unroll
    for (int j = 0; j < 8; ++j) {
      float val = c[j];
      if (HAS_RESID) val += pr[j*32];
      po[j*32] = val;
    }
  }
}

static void launch_gemm(const float* in, const float* W, const float* bias,
                        const float* resid, float* out, long M, int kd, hipStream_t st) {
  int blocks = cdiv(M, 128);
  if (kd == 16) {
    k_gemm32<16,false,false><<<blocks, 256, 0, st>>>(in, W, nullptr, nullptr, out, (int)M);
  } else if (bias && resid) {
    k_gemm32<32,true, true ><<<blocks, 256, 0, st>>>(in, W, bias, resid, out, (int)M);
  } else if (bias) {
    k_gemm32<32,true, false><<<blocks, 256, 0, st>>>(in, W, bias, nullptr, out, (int)M);
  } else if (resid) {
    k_gemm32<32,false,true ><<<blocks, 256, 0, st>>>(in, W, nullptr, resid, out, (int)M);
  } else {
    k_gemm32<32,false,false><<<blocks, 256, 0, st>>>(in, W, nullptr, nullptr, out, (int)M);
  }
}

// ---------------- full equivariant message pass (one thread per (edge, f)) ----------------
__global__ void k_message(const float* __restrict__ s, const float* __restrict__ v,
                          const float* __restrict__ ps, const float* __restrict__ pv,
                          const float* __restrict__ pb0, const float* __restrict__ pb1,
                          const float* __restrict__ pw,   // (10, F)
                          const int* __restrict__ dst, const int* __restrict__ src,
                          float* __restrict__ ag_s, float* __restrict__ ag_v,
                          float* __restrict__ ag_ps, float* __restrict__ ag_pv) {
  long idx = (long)blockIdx.x * blockDim.x + threadIdx.x;
  if (idx >= (long)EE * FF) return;
  int e = (int)(idx >> 5), f = (int)(idx & 31);
  int sn = src[e], dn = dst[e];
  if (e + 64 < EE) {
    __builtin_prefetch(&pb0[(long)(e + 64)*FF], 0, 0);   // global_prefetch_b8
  }
  float se  = s [sn*FF + f];
  float pse = ps[sn*FF + f];
  float ve0 = v [(sn*3+0)*FF + f], ve1 = v [(sn*3+1)*FF + f], ve2 = v [(sn*3+2)*FF + f];
  float pe0 = pv[(sn*3+0)*FF + f], pe1 = pv[(sn*3+1)*FF + f], pe2 = pv[(sn*3+2)*FF + f];
  float p0  = pb0[(long)e*FF + f];
  float q0  = pb1[((long)e*3+0)*FF + f];
  float q1  = pb1[((long)e*3+1)*FF + f];
  float q2  = pb1[((long)e*3+2)*FF + f];

  float w0 = pw[0*FF+f], w1 = pw[1*FF+f], w2 = pw[2*FF+f], w3 = pw[3*FF+f];
  float w4 = pw[4*FF+f], w5 = pw[5*FF+f], w6 = pw[6*FF+f], w7 = pw[7*FF+f];
  float w8 = pw[8*FF+f], w9 = pw[9*FF+f];

  float m_s  = w0*se*p0  + w1*(ve0*q0 + ve1*q1 + ve2*q2);
  float m_ps = w2*pse*p0 + w3*(pe0*q0 + pe1*q1 + pe2*q2);
  float cx0 = pe1*q2 - pe2*q1, cx1 = pe2*q0 - pe0*q2, cx2 = pe0*q1 - pe1*q0;  // cross(pve,pb1)
  float dx0 = ve1*q2 - ve2*q1, dx1 = ve2*q0 - ve0*q2, dx2 = ve0*q1 - ve1*q0;  // cross(ve,pb1)
  float m_v0 = w4*ve0*p0 + w5*se*q0 + w6*cx0;
  float m_v1 = w4*ve1*p0 + w5*se*q1 + w6*cx1;
  float m_v2 = w4*ve2*p0 + w5*se*q2 + w6*cx2;
  float m_p0 = w7*pe0*p0 + w8*pse*q0 + w9*dx0;
  float m_p1 = w7*pe1*p0 + w8*pse*q1 + w9*dx1;
  float m_p2 = w7*pe2*p0 + w8*pse*q2 + w9*dx2;

  fatomic(&ag_s [dn*FF + f],        m_s);
  fatomic(&ag_ps[dn*FF + f],        m_ps);
  fatomic(&ag_v [(dn*3+0)*FF + f],  m_v0);
  fatomic(&ag_v [(dn*3+1)*FF + f],  m_v1);
  fatomic(&ag_v [(dn*3+2)*FF + f],  m_v2);
  fatomic(&ag_pv[(dn*3+0)*FF + f],  m_p0);
  fatomic(&ag_pv[(dn*3+1)*FF + f],  m_p1);
  fatomic(&ag_pv[(dn*3+2)*FF + f],  m_p2);
}

// ---------------- scalar-only last message pass ----------------
__global__ void k_message_last(const float* __restrict__ s, const float* __restrict__ v,
                               const float* __restrict__ pb0, const float* __restrict__ pb1,
                               const float* __restrict__ pwl,  // (2, F)
                               const int* __restrict__ dst, const int* __restrict__ src,
                               float* __restrict__ ag_s) {
  long idx = (long)blockIdx.x * blockDim.x + threadIdx.x;
  if (idx >= (long)EE * FF) return;
  int e = (int)(idx >> 5), f = (int)(idx & 31);
  int sn = src[e], dn = dst[e];
  float se  = s[sn*FF + f];
  float ve0 = v[(sn*3+0)*FF + f], ve1 = v[(sn*3+1)*FF + f], ve2 = v[(sn*3+2)*FF + f];
  float p0  = pb0[(long)e*FF + f];
  float q0  = pb1[((long)e*3+0)*FF + f];
  float q1  = pb1[((long)e*3+1)*FF + f];
  float q2  = pb1[((long)e*3+2)*FF + f];
  float m = pwl[0*FF+f]*se*p0 + pwl[1*FF+f]*(ve0*q0 + ve1*q1 + ve2*q2);
  fatomic(&ag_s[dn*FF + f], m);
}

// ---------------- gated silu across channels ----------------
__global__ void k_gate(float* __restrict__ ys, float* __restrict__ yv,
                       float* __restrict__ yps, float* __restrict__ ypv) {
  long i = (long)blockIdx.x * blockDim.x + threadIdx.x;
  if (i >= (long)NN * FF) return;
  int n = (int)(i >> 5), f = (int)(i & 31);
  float x = ys[i];
  float g = 1.f / (1.f + expf(-x));
  ys[i]  = x * g;
  yps[i] *= g;
  #pragma unroll
  for (int c = 0; c < 3; ++c) {
    yv [(n*3+c)*FF + f] *= g;
    ypv[(n*3+c)*FF + f] *= g;
  }
}

__global__ void k_silu(float* __restrict__ y, long n) {
  long i = (long)blockIdx.x * blockDim.x + threadIdx.x;
  if (i >= n) return;
  float x = y[i];
  y[i] = x / (1.f + expf(-x));
}

// ---------------- readout: atomic energies -> per-graph energy ----------------
__global__ void k_energy(const float* __restrict__ s, const float* __restrict__ w_out,
                         const float* __restrict__ eb, const int* __restrict__ Z,
                         const int* __restrict__ bseg, float* __restrict__ energy) {
  int n = blockIdx.x * blockDim.x + threadIdx.x;
  if (n >= NN) return;
  float e = eb[Z[n]];
  #pragma unroll
  for (int f = 0; f < FF; ++f) e += s[n*FF + f] * w_out[f];
  fatomic(&energy[bseg[n]], e);
}

// ---------------- forces: analytic gradient through the last-pass radial basis ----------
__global__ void k_forces(const float* __restrict__ s, const float* __restrict__ v,
                         const float* __restrict__ uvec, const float* __restrict__ rvec,
                         const float* __restrict__ Wb0l, const float* __restrict__ Wb1l,
                         const float* __restrict__ pwl, const float* __restrict__ w_out,
                         const int* __restrict__ dst, const int* __restrict__ src,
                         float* __restrict__ forces) {
  int e = blockIdx.x * blockDim.x + threadIdx.x;
  if (e >= EE) return;
  int sn = src[e], dn = dst[e];
  float r  = rvec[e];
  float u0 = uvec[e*3+0], u1 = uvec[e*3+1], u2 = uvec[e*3+2];
  float t = r / (1.f + r);
  float dtdr = 1.f / ((1.f + r) * (1.f + r));
  float omt = 1.f - t;
  float cut = 0.f, cutd = 0.f;
  if (r < CUTR) {
    float q = (CUTR - r) * (CUTR + r);
    cut  = expf(-r * r / q);
    cutd = cut * (-2.f * r * CUTR * CUTR / (q * q));
  }
  float o[16];
  o[0] = 1.f;
  #pragma unroll
  for (int j = 1; j < 16; ++j) o[j] = o[j-1] * omt;

  float rad[16], radp[16];
  {
    float tkm1 = 0.f, tk = 1.f;
    #pragma unroll
    for (int k = 0; k < 16; ++k) {
      float base  = c_binom[k] * tk * o[15-k];
      float term1 = (k > 0)  ? (float)k * tkm1 * o[15-k]    : 0.f;
      float term2 = (k < 15) ? (float)(15-k) * tk * o[14-k] : 0.f;
      float basep = c_binom[k] * dtdr * (term1 - term2);
      rad[k]  = base * cut;
      radp[k] = basep * cut + base * cutd;
      tkm1 = tk; tk *= t;
    }
  }

  float alpha[16], beta[16];
  #pragma unroll
  for (int k = 0; k < 16; ++k) { alpha[k] = 0.f; beta[k] = 0.f; }
  float dEdu0 = 0.f, dEdu1 = 0.f, dEdu2 = 0.f;

  for (int f = 0; f < FF; ++f) {
    float sf  = s[sn*FF + f];
    float v0f = v[(sn*3+0)*FF + f];
    float v1f = v[(sn*3+1)*FF + f];
    float v2f = v[(sn*3+2)*FF + f];
    float wo  = w_out[f];
    float P   = wo * pwl[0*FF+f] * sf;
    float Q   = wo * pwl[1*FF+f];
    float Vu  = v0f*u0 + v1f*u1 + v2f*u2;
    float Rf  = 0.f;
    #pragma unroll
    for (int k = 0; k < 16; ++k) {
      float wb0 = Wb0l[k*FF + f];
      float wb1 = Wb1l[k*FF + f];
      alpha[k] += P * wb0;
      beta[k]  += Q * Vu * wb1;
      Rf       += rad[k] * wb1;
    }
    dEdu0 += Q * v0f * Rf;
    dEdu1 += Q * v1f * Rf;
    dEdu2 += Q * v2f * Rf;
  }

  float dEdr = 0.f;
  #pragma unroll
  for (int k = 0; k < 16; ++k) dEdr += radp[k] * (alpha[k] + beta[k]);

  float udot = dEdu0*u0 + dEdu1*u1 + dEdu2*u2;
  float inv_r = 1.f / r;
  float g0 = dEdr*u0 + (dEdu0 - udot*u0) * inv_r;
  float g1 = dEdr*u1 + (dEdu1 - udot*u1) * inv_r;
  float g2 = dEdr*u2 + (dEdu2 - udot*u2) * inv_r;

  fatomic(&forces[sn*3+0], -g0);
  fatomic(&forces[sn*3+1], -g1);
  fatomic(&forces[sn*3+2], -g2);
  fatomic(&forces[dn*3+0],  g0);
  fatomic(&forces[dn*3+1],  g1);
  fatomic(&forces[dn*3+2],  g2);
}

// ==================================================================================
extern "C" void kernel_launch(void* const* d_in, const int* in_sizes, int n_in,
                              void* d_out, int out_size, void* d_ws, size_t ws_size,
                              hipStream_t stream) {
  (void)in_sizes; (void)n_in; (void)out_size; (void)ws_size;
  const int*   Z     = (const int*)  d_in[0];
  const float* pos   = (const float*)d_in[1];
  const int*   dsti  = (const int*)  d_in[2];
  const int*   srci  = (const int*)  d_in[3];
  const int*   bseg  = (const int*)  d_in[4];
  const float* embed = (const float*)d_in[6];
  const float* Wb0   = (const float*)d_in[7];
  const float* Wb1   = (const float*)d_in[8];
  const float* pw    = (const float*)d_in[9];
  const float* Wd1   = (const float*)d_in[10];
  const float* bd1   = (const float*)d_in[11];
  const float* Wd2   = (const float*)d_in[12];
  const float* bd2   = (const float*)d_in[13];
  const float* Wb0l  = (const float*)d_in[14];
  const float* Wb1l  = (const float*)d_in[15];
  const float* pwl   = (const float*)d_in[16];
  const float* Wd1l  = (const float*)d_in[17];
  const float* bd1l  = (const float*)d_in[18];
  const float* Wd2l  = (const float*)d_in[19];
  const float* bd2l  = (const float*)d_in[20];
  const float* w_out = (const float*)d_in[21];
  const float* ebias = (const float*)d_in[22];

  float* ws = (float*)d_ws;
  size_t off = 0;
  float* b0   = ws + off; off += (size_t)EE * KK;
  float* b1   = ws + off; off += (size_t)EE * 3 * KK;
  float* uvec = ws + off; off += (size_t)EE * 3;
  float* rvec = ws + off; off += (size_t)EE;
  float* pb0  = ws + off; off += (size_t)EE * FF;
  float* pb1  = ws + off; off += (size_t)EE * 3 * FF;
  float* st   = ws + off; off += (size_t)NN * 8 * FF;   // s | v | ps | pv
  float* ag   = ws + off; off += (size_t)NN * 8 * FF;
  float* yb   = ws + off; off += (size_t)NN * 8 * FF;
  float* s_  = st;                    float* v_  = st + (size_t)NN*FF;
  float* ps_ = v_ + (size_t)3*NN*FF;  float* pv_ = ps_ + (size_t)NN*FF;
  float* as_ = ag;                    float* av_ = ag + (size_t)NN*FF;
  float* aps = av_ + (size_t)3*NN*FF; float* apv = aps + (size_t)NN*FF;
  float* ys_ = yb;                    float* yv_ = yb + (size_t)NN*FF;
  float* yps = yv_ + (size_t)3*NN*FF; float* ypv = yps + (size_t)NN*FF;

  float* energy_out = (float*)d_out;
  float* forces_out = (float*)d_out + BB;

  const int T = 256;
  const long stateN = (long)NN * 8 * FF;

  // init output + state
  k_zero<<<cdiv(BB + (long)3*NN, T), T, 0, stream>>>(energy_out, BB + (long)3*NN);
  k_zero<<<cdiv(stateN, T), T, 0, stream>>>(st, stateN);
  k_embed<<<cdiv((long)NN*FF, T), T, 0, stream>>>(Z, embed, s_);
  k_basis<<<cdiv(EE, T), T, 0, stream>>>(pos, dsti, srci, b0, b1, uvec, rvec);

  for (int i = 0; i < 4; ++i) {
    // project radial basis to feature space (WMMA, K=16 zero-padded)
    launch_gemm(b0, Wb0 + (size_t)i*KK*FF, nullptr, nullptr, pb0, EE, 16, stream);
    launch_gemm(b1, Wb1 + (size_t)i*KK*FF, nullptr, nullptr, pb1, (long)3*EE, 16, stream);
    // message pass + segment sum
    k_zero<<<cdiv(stateN, T), T, 0, stream>>>(ag, stateN);
    k_message<<<cdiv((long)EE*FF, T), T, 0, stream>>>(s_, v_, ps_, pv_, pb0, pb1,
        pw + (size_t)i*10*FF, dsti, srci, as_, av_, aps, apv);
    // y = state + aggregate (stored into ag)
    k_add_inplace<<<cdiv(stateN, T), T, 0, stream>>>(ag, st, stateN);
    // dense-1 (WMMA)
    const float* W1 = Wd1 + (size_t)i*4*FF*FF;
    launch_gemm(as_, W1 + 0*FF*FF, bd1 + (size_t)i*FF, nullptr, ys_, NN, 32, stream);
    launch_gemm(av_, W1 + 1*FF*FF, nullptr, nullptr, yv_, (long)3*NN, 32, stream);
    launch_gemm(aps, W1 + 2*FF*FF, nullptr, nullptr, yps, NN, 32, stream);
    launch_gemm(apv, W1 + 3*FF*FF, nullptr, nullptr, ypv, (long)3*NN, 32, stream);
    // gated silu
    k_gate<<<cdiv((long)NN*FF, T), T, 0, stream>>>(ys_, yv_, yps, ypv);
    // dense-2 (WMMA) + residual into state
    const float* W2 = Wd2 + (size_t)i*4*FF*FF;
    launch_gemm(ys_, W2 + 0*FF*FF, bd2 + (size_t)i*FF, s_, s_, NN, 32, stream);
    launch_gemm(yv_, W2 + 1*FF*FF, nullptr, v_, v_, (long)3*NN, 32, stream);
    launch_gemm(yps, W2 + 2*FF*FF, nullptr, ps_, ps_, NN, 32, stream);
    launch_gemm(ypv, W2 + 3*FF*FF, nullptr, pv_, pv_, (long)3*NN, 32, stream);
  }

  // ---- last scalar-only pass ----
  launch_gemm(b0, Wb0l, nullptr, nullptr, pb0, EE, 16, stream);
  launch_gemm(b1, Wb1l, nullptr, nullptr, pb1, (long)3*EE, 16, stream);
  k_zero<<<cdiv((long)NN*FF, T), T, 0, stream>>>(as_, (long)NN*FF);
  k_message_last<<<cdiv((long)EE*FF, T), T, 0, stream>>>(s_, v_, pb0, pb1, pwl, dsti, srci, as_);
  k_add_inplace<<<cdiv((long)NN*FF, T), T, 0, stream>>>(as_, s_, (long)NN*FF);
  launch_gemm(as_, Wd1l, bd1l, nullptr, ys_, NN, 32, stream);
  k_silu<<<cdiv((long)NN*FF, T), T, 0, stream>>>(ys_, (long)NN*FF);
  launch_gemm(ys_, Wd2l, bd2l, s_, s_, NN, 32, stream);

  // ---- readout + forces ----
  k_energy<<<cdiv(NN, T), T, 0, stream>>>(s_, w_out, ebias, Z, bseg, energy_out);
  k_forces<<<cdiv(EE, T), T, 0, stream>>>(s_, v_, uvec, rvec, Wb0l, Wb1l, pwl, w_out,
                                          dsti, srci, forces_out);
}
